// NetVLAD_35098472742931
// MI455X (gfx1250) — matrix operand order repaired and checked by reading, not compiled, not run on hardware
//
#include <hip/hip_runtime.h>

typedef __bf16 bf16_t;
typedef __attribute__((ext_vector_type(16))) __bf16 v16bf;
typedef __attribute__((ext_vector_type(8)))  __bf16 v8bf;
typedef __attribute__((ext_vector_type(8)))  float  v8f;

#define B_    32
#define N_    8192
#define D_    128
#define K_    64
#define SPLIT 16                      // workgroups per batch
#define ROWS  64                      // rows per chunk
#define CHUNKS (N_ / SPLIT / ROWS)    // 8 chunks per workgroup

// ---- LDS layout (bytes). xa (bf16[64][128]) and L (f32[64][68]) overlap:
// xa lives normalize->GEMM1, L lives GEMM1-writeback->softmax (disjoint).
#define XA_OFF   0
#define L_OFF    0
#define R0_SZ    17408                       // max(16384, 64*68*4)
#define XT_OFF   R0_SZ                       // bf16[128][72] = 18432
#define AT_OFF   (XT_OFF + 18432)            // bf16[64][72]  = 9216
#define WB_OFF   (AT_OFF + 9216)             // bf16[64][128] = 16384
#define BIAS_OFF (WB_OFF + 16384)            // f32[64]       = 256
#define SMEM_SZ  (BIAS_OFF + 256)            // 61696 < 64KB

// Build a v16bf WMMA operand from two contiguous 16-byte LDS chunks.
__device__ inline v16bf ld16(const bf16_t* lo, const bf16_t* hi) {
  v8bf a = *reinterpret_cast<const v8bf*>(lo);
  v8bf b = *reinterpret_cast<const v8bf*>(hi);
  v16bf r;
#pragma unroll
  for (int i = 0; i < 8; ++i) { r[i] = a[i]; r[i + 8] = b[i]; }
  return r;
}

__device__ inline void load_rows(const float* xp, float4 (&v)[8]) {
#pragma unroll
  for (int j = 0; j < 8; ++j) v[j] = reinterpret_cast<const float4*>(xp)[j];
}

__global__ __launch_bounds__(256) void netvlad_assign_accum(
    const float* __restrict__ x, const float* __restrict__ conv_w,
    const float* __restrict__ conv_b, float* __restrict__ gax,
    float* __restrict__ gasum)
{
  __shared__ __align__(16) char smem[SMEM_SZ];
  bf16_t* xa  = reinterpret_cast<bf16_t*>(smem + XA_OFF);
  float*  L   = reinterpret_cast<float* >(smem + L_OFF);
  bf16_t* xt  = reinterpret_cast<bf16_t*>(smem + XT_OFF);
  bf16_t* at  = reinterpret_cast<bf16_t*>(smem + AT_OFF);
  bf16_t* wB  = reinterpret_cast<bf16_t*>(smem + WB_OFF);
  float*  bia = reinterpret_cast<float* >(smem + BIAS_OFF);

  const int tid    = threadIdx.x;
  const int wave   = tid >> 5;
  const int lane   = tid & 31;       // wave32
  const int lane16 = lane & 15;
  const int laneHi = lane >> 4;
  const int b = blockIdx.y;
  const int s = blockIdx.x;

  // one-time: conv_w -> bf16 LDS, bias -> LDS
  for (int i = tid; i < K_ * D_; i += 256) wB[i] = (bf16_t)conv_w[i];
  if (tid < K_) bia[tid] = conv_b[tid];

  // GEMM2 accumulators persist across all chunks: 4 tiles x 8 f32 per lane
  v8f acc2[4];
#pragma unroll
  for (int t = 0; t < 4; ++t)
#pragma unroll
    for (int i = 0; i < 8; ++i) acc2[t][i] = 0.f;
  float asum_acc = 0.f;

  const int row = tid >> 2;  // 0..63
  const int sub = tid & 3;   // 0..3 (32 dims each)

  // software pipeline: this thread's global pointer, chunk 0 preloaded
  const float* xp = x + ((size_t)b * N_ + s * (ROWS * CHUNKS) + row) * D_ + sub * 32;
  float4 buf[8];
  load_rows(xp, buf);

  for (int c = 0; c < CHUNKS; ++c) {
    __syncthreads();  // xa/L, xt, at free for reuse

    // ---- L2-normalize 64 rows (consumes pipelined register buffer) ----
    {
      float ss = 0.f;
#pragma unroll
      for (int j = 0; j < 8; ++j)
        ss += buf[j].x * buf[j].x + buf[j].y * buf[j].y +
              buf[j].z * buf[j].z + buf[j].w * buf[j].w;
      ss += __shfl_xor(ss, 1);
      ss += __shfl_xor(ss, 2);
      const float inv = 1.f / fmaxf(sqrtf(ss), 1e-12f);
#pragma unroll
      for (int j = 0; j < 8; ++j) {
        const int d0 = sub * 32 + j * 4;
        float vals[4] = { buf[j].x, buf[j].y, buf[j].z, buf[j].w };
#pragma unroll
        for (int q = 0; q < 4; ++q) {
          bf16_t h = (bf16_t)(vals[q] * inv);
          xa[row * 128 + d0 + q]  = h;   // row-major  (GEMM1 A)
          xt[(d0 + q) * 72 + row] = h;   // transposed (GEMM2 B)
        }
      }
    }
    // issue next chunk's loads now; they drain during GEMM1/softmax/GEMM2
    if (c + 1 < CHUNKS) {
      load_rows(xp + (size_t)(c + 1) * ROWS * D_, buf);
      if (c + 2 < CHUNKS)
        __builtin_prefetch(xp + (size_t)(c + 2) * ROWS * D_, 0, 3);
    }
    __syncthreads();

    // ---- GEMM1: logits[64 rows][64 clusters] = xa * wB^T ----
    v8f c1[2];
#pragma unroll
    for (int t = 0; t < 2; ++t) {
#pragma unroll
      for (int i = 0; i < 8; ++i) c1[t][i] = 0.f;
      const int tile = wave * 2 + t;
      const int rt = tile >> 2, ct = tile & 3;
      const bf16_t* arow = xa + (rt * 16 + lane16) * 128;
      const bf16_t* brow = wB + (ct * 16 + lane16) * 128;
#pragma unroll
      for (int kb = 0; kb < 128; kb += 32) {
        // A: lane m holds A[m][k..k+7] and A[m][k+16..k+23] (hi lanes +8)
        v16bf A  = ld16(arow + kb + laneHi * 8, arow + kb + 16 + laneHi * 8);
        // B: lane n holds column n, K k..k+7 and k+16..k+23 (hi lanes +8)
        v16bf Bv = ld16(brow + kb + laneHi * 8, brow + kb + 16 + laneHi * 8);
        c1[t] = __builtin_amdgcn_wmma_f32_16x16x32_bf16(false, A, false, Bv,
                                                        (short)0, c1[t], false, false);
      }
    }
    __syncthreads();  // all waves done reading xa; L may overwrite it
#pragma unroll
    for (int t = 0; t < 2; ++t) {
      const int tile = wave * 2 + t;
      const int rt = tile >> 2, ct = tile & 3;
      const int r0 = rt * 16 + laneHi * 8;
      const int col = ct * 16 + lane16;
#pragma unroll
      for (int r = 0; r < 8; ++r) L[(r0 + r) * 68 + col] = c1[t][r];
    }
    __syncthreads();

    // ---- softmax over 64 clusters (4 lanes per row) ----
    {
      const float* Lr = L + row * 68;
      float e[16];
      float m = -3.4e38f;
#pragma unroll
      for (int j = 0; j < 16; ++j) {
        e[j] = Lr[sub * 16 + j] + bia[sub * 16 + j];
        m = fmaxf(m, e[j]);
      }
      m = fmaxf(m, __shfl_xor(m, 1));
      m = fmaxf(m, __shfl_xor(m, 2));
      float ssum = 0.f;
#pragma unroll
      for (int j = 0; j < 16; ++j) { e[j] = __expf(e[j] - m); ssum += e[j]; }
      ssum += __shfl_xor(ssum, 1);
      ssum += __shfl_xor(ssum, 2);
      const float inv = 1.f / ssum;
#pragma unroll
      for (int j = 0; j < 16; ++j)
        at[(sub * 16 + j) * 72 + row] = (bf16_t)(e[j] * inv);  // transposed
    }
    __syncthreads();

    // ---- a_sum column accumulation ----
    if (tid < K_) {
      const bf16_t* ar = at + tid * 72;
      float a = 0.f;
#pragma unroll
      for (int r = 0; r < ROWS; ++r) a += (float)ar[r];
      asum_acc += a;
    }

    // ---- GEMM2: ax[64 clusters][128 dims] += a^T * xn ----
#pragma unroll
    for (int t = 0; t < 4; ++t) {
      const int tile = wave * 4 + t;
      const int ct = tile >> 3, dt = tile & 7;
      const bf16_t* arow = at + (ct * 16 + lane16) * 72;
      const bf16_t* brow = xt + (dt * 16 + lane16) * 72;
#pragma unroll
      for (int kb = 0; kb < ROWS; kb += 32) {
        v16bf A  = ld16(arow + kb + laneHi * 8, arow + kb + 16 + laneHi * 8);
        v16bf Bv = ld16(brow + kb + laneHi * 8, brow + kb + 16 + laneHi * 8);
        acc2[t] = __builtin_amdgcn_wmma_f32_16x16x32_bf16(false, A, false, Bv,
                                                          (short)0, acc2[t], false, false);
      }
    }
  }

  __syncthreads();
  // merge register-resident partials into per-batch global accumulators
#pragma unroll
  for (int t = 0; t < 4; ++t) {
    const int tile = wave * 4 + t;
    const int ct = tile >> 3, dt = tile & 7;
    const int r0 = ct * 16 + laneHi * 8;     // cluster
    const int col = dt * 16 + lane16;        // dim
#pragma unroll
    for (int r = 0; r < 8; ++r)
      atomicAdd(&gax[((size_t)b * K_ + r0 + r) * D_ + col], acc2[t][r]);
  }
  if (tid < K_) atomicAdd(&gasum[b * K_ + tid], asum_acc);
}

__global__ __launch_bounds__(128) void netvlad_finalize(
    const float* __restrict__ centroids, const float* __restrict__ gax,
    const float* __restrict__ gasum, float* __restrict__ gnorm,
    float* __restrict__ out)
{
  const int bk = blockIdx.x;            // b*64 + k
  const int b = bk >> 6, k = bk & 63;
  const int d = threadIdx.x;
  const float asum = gasum[b * K_ + k];
  const float v = gax[(size_t)bk * D_ + d] - centroids[k * D_ + d] * asum;
  float t = v * v;
#pragma unroll
  for (int off = 16; off > 0; off >>= 1) t += __shfl_xor(t, off);
  __shared__ float red[4];
  if ((threadIdx.x & 31) == 0) red[threadIdx.x >> 5] = t;
  __syncthreads();
  const float total = red[0] + red[1] + red[2] + red[3];
  const float inv = 1.f / fmaxf(sqrtf(total), 1e-12f);
  out[(size_t)bk * D_ + d] = v * inv;
  if (threadIdx.x == 0) atomicAdd(&gnorm[b], total * inv * inv);
}

__global__ __launch_bounds__(256) void netvlad_gnorm(
    const float* __restrict__ gnorm, float* __restrict__ out)
{
  const int b = blockIdx.x;
  const float inv = 1.f / fmaxf(sqrtf(gnorm[b]), 1e-12f);
  for (int i = threadIdx.x; i < K_ * D_; i += 256)
    out[(size_t)b * K_ * D_ + i] *= inv;
}

extern "C" void kernel_launch(void* const* d_in, const int* in_sizes, int n_in,
                              void* d_out, int out_size, void* d_ws, size_t ws_size,
                              hipStream_t stream) {
  (void)in_sizes; (void)n_in; (void)out_size; (void)ws_size;
  const float* x         = (const float*)d_in[0];
  const float* centroids = (const float*)d_in[1];
  const float* conv_w    = (const float*)d_in[2];
  const float* conv_b    = (const float*)d_in[3];
  float* out = (float*)d_out;

  float* gax   = (float*)d_ws;                         // [B,K,D]
  float* gasum = gax + (size_t)B_ * K_ * D_;           // [B,K]
  float* gnorm = gasum + (size_t)B_ * K_;              // [B]
  const size_t zbytes = ((size_t)B_ * K_ * D_ + B_ * K_ + B_) * sizeof(float);
  hipMemsetAsync(d_ws, 0, zbytes, stream);

  netvlad_assign_accum<<<dim3(SPLIT, B_), 256, 0, stream>>>(x, conv_w, conv_b, gax, gasum);
  netvlad_finalize<<<B_ * K_, 128, 0, stream>>>(centroids, gax, gasum, gnorm, out);
  netvlad_gnorm<<<B_, 256, 0, stream>>>(gnorm, out);
}
